// SessionGraph_36172214567706
// MI455X (gfx1250) — matrix operand order repaired and checked by reading, not compiled, not run on hardware
//
#include <hip/hip_runtime.h>
#include <hip/hip_bf16.h>

typedef __attribute__((ext_vector_type(16))) __bf16 v16bf;
typedef __attribute__((ext_vector_type(8)))  float  v8f;

#define B_SZ   1024
#define L_SZ   50
#define D_SZ   128
#define FF_SZ  256
#define NNODE  50000
#define NNZ_SZ 800000
#define NTOK   (B_SZ * L_SZ)   // 51200

// ---------------------------------------------------------------------------
__device__ __forceinline__ unsigned short f32_to_bf16u(float f) {
    union { float f; unsigned int u; } x; x.f = f;
    unsigned int u = x.u + 0x7FFFu + ((x.u >> 16) & 1u);
    return (unsigned short)(u >> 16);
}
__device__ __forceinline__ unsigned pack_bf16(float lo, float hi) {
    return (unsigned)f32_to_bf16u(lo) | ((unsigned)f32_to_bf16u(hi) << 16);
}

// CDNA5 async global->LDS copy (ASYNCcnt-tracked), 16 bytes per lane.
__device__ __forceinline__ void async_load_b128(unsigned lds_byte_off, const void* gaddr) {
    asm volatile("global_load_async_to_lds_b128 %0, %1, off"
                 :: "v"(lds_byte_off), "v"(gaddr) : "memory");
}
__device__ __forceinline__ void wait_async0() {
#if __has_builtin(__builtin_amdgcn_s_wait_asynccnt)
    __builtin_amdgcn_s_wait_asynccnt(0);
#else
    asm volatile("s_wait_asynccnt 0x0" ::: "memory");
#endif
}

// ---------------------------------------------------------------------------
// fp32 -> packed bf16 (weight mirrors; halves weight bandwidth, feeds async path)
__global__ void sg_f32_to_bf16_kernel(const float* __restrict__ in,
                                      unsigned short* __restrict__ op, long n)
{
    long i = ((long)blockIdx.x * blockDim.x + threadIdx.x) * 8;
    if (i >= n) return;
    if (i + 8 <= n) {
        float4 a = *(const float4*)&in[i];
        float4 b = *(const float4*)&in[i + 4];
        uint4 w;
        w.x = pack_bf16(a.x, a.y); w.y = pack_bf16(a.z, a.w);
        w.z = pack_bf16(b.x, b.y); w.w = pack_bf16(b.z, b.w);
        *(uint4*)&op[i] = w;
    } else {
        for (long j = i; j < n; ++j) op[j] = f32_to_bf16u(in[j]);
    }
}

// ---------------------------------------------------------------------------
// GEMM: C/Cb[M,N] = act( Ab[M,K](lda,bf16) @ Bb[N,K]^T(bf16) + bias + resid )
// Block 256 thr = 8 waves; block tile 64x64; wave tile 16x32 (2 WMMA/stage).
// BOTH A and B tiles double-buffered via global_load_async_to_lds_b128:
// inner loop = 2 async issues + ds_load_b128 fragments + 2 v_wmma. No cvt.
#define BM 64
#define BN 64
#define BK 32
#define LT_STR 40   // ushorts per tile row (32 data + 8 pad) -> 80B, 16B-aligned

__global__ __launch_bounds__(256)
void sg_gemm_bf16_wmma(const unsigned short* __restrict__ Ab, int lda,
                       const unsigned short* __restrict__ Bb,
                       const float* __restrict__ bias,
                       const float* __restrict__ resid,
                       float* __restrict__ C,
                       unsigned short* __restrict__ Cb,
                       int M, int N, int K, int act)
{
    __shared__ __align__(16) unsigned short la[2][BM * LT_STR];
    __shared__ __align__(16) unsigned short lb[2][BN * LT_STR];

    const int tid  = threadIdx.x;
    const int wave = tid >> 5;
    const int lane = tid & 31;
    const int wm   = wave >> 1;      // 0..3 -> M quarter
    const int wn   = wave & 1;       // 0..1 -> N half
    const int m0 = blockIdx.y * BM;
    const int n0 = blockIdx.x * BN;

    // async slot: row r (0..63), 16B segment seg (ushort units); 256 lanes = 4KB tile
    const int r   = tid >> 2;
    const int seg = (tid & 3) * 8;
    const int gmA = min(m0 + r, M - 1);           // clamp: branch-free, garbage rows
    const int gnB = min(n0 + r, N - 1);           //   never stored by epilogue
    const unsigned short* aSeg = Ab + (long)gmA * lda + seg;
    const unsigned short* bSeg = Bb + (long)gnB * K + seg;
    const unsigned laBase = (unsigned)(size_t)&la[0][0];
    const unsigned lbBase = (unsigned)(size_t)&lb[0][0];
    const unsigned rowOff = (unsigned)(r * LT_STR + seg) * 2u;
    const unsigned bufB   = (unsigned)(BM * LT_STR) * 2u;

    const int ln  = lane & 15;
    const int kb  = (lane >> 4) * 8;   // A-fragment K sub-base (ISA 7.12.2 A layout)
    const int khB = (lane >> 4) * 16;  // unused: B fragment uses same K order as A (kb)
    v8f acc0 = {}, acc1 = {};
    const int T = K / BK;

    // ---- prologue: stage 0 ----
    async_load_b128(laBase + rowOff, aSeg);
    async_load_b128(lbBase + rowOff, bSeg);
    wait_async0();
    __syncthreads();

    for (int t = 0; t < T; ++t) {
        const int cbuf = t & 1;
        const int nbuf = cbuf ^ 1;
        const bool hasNext = (t + 1) < T;
        if (hasNext) {
            __builtin_prefetch(aSeg + (t + 1) * BK + BK, 0, 0);  // global_prefetch_b8
            async_load_b128(laBase + nbuf * bufB + rowOff, aSeg + (long)(t + 1) * BK);
            async_load_b128(lbBase + nbuf * bufB + rowOff, bSeg + (long)(t + 1) * BK);
        }
        // fragments
        union { v16bf v; unsigned int d[8]; } fa, fb0, fb1;
        const unsigned short* pa  = &la[cbuf][(wm * 16 + ln) * LT_STR];
        const unsigned short* pb0 = &lb[cbuf][(wn * 32 + ln) * LT_STR];
        const unsigned short* pb1 = pb0 + 16 * LT_STR;
#pragma unroll
        for (int q = 0; q < 8; ++q) {
            int kk = ((q >= 4) ? 16 : 0) + kb + (q & 3) * 2;
            fa.d[q]  = *(const unsigned int*)&pa[kk];
            fb0.d[q] = *(const unsigned int*)&pb0[kk];
            fb1.d[q] = *(const unsigned int*)&pb1[kk];
        }
        acc0 = __builtin_amdgcn_wmma_f32_16x16x32_bf16(
                   false, fa.v, false, fb0.v, (short)0, acc0, false, false);
        acc1 = __builtin_amdgcn_wmma_f32_16x16x32_bf16(
                   false, fa.v, false, fb1.v, (short)0, acc1, false, false);
        __syncthreads();
        if (hasNext) {
            wait_async0();
            __syncthreads();
        }
    }

    // ---- epilogue ----
    const int cn0 = n0 + wn * 32 + ln;
    const int cn1 = cn0 + 16;
    const int mb  = m0 + wm * 16 + ((lane >= 16) ? 8 : 0);
    const float bs0 = (bias && cn0 < N) ? bias[cn0] : 0.0f;
    const float bs1 = (bias && cn1 < N) ? bias[cn1] : 0.0f;
#pragma unroll
    for (int q = 0; q < 8; ++q) {
        int gm = mb + q;
        if (gm < M) {
            if (cn0 < N) {
                float v = acc0[q] + bs0;
                if (resid) v += resid[(long)gm * N + cn0];
                if (act == 1) v = fmaxf(v, 0.0f);
                if (C)  C [(long)gm * N + cn0] = v;
                if (Cb) Cb[(long)gm * N + cn0] = f32_to_bf16u(v);
            }
            if (cn1 < N) {
                float v = acc1[q] + bs1;
                if (resid) v += resid[(long)gm * N + cn1];
                if (act == 1) v = fmaxf(v, 0.0f);
                if (C)  C [(long)gm * N + cn1] = v;
                if (Cb) Cb[(long)gm * N + cn1] = f32_to_bf16u(v);
            }
        }
    }
}

// ---------------------------------------------------------------------------
__global__ void sg_zero_kernel(float* p, long n) {
    long i = (long)blockIdx.x * blockDim.x + threadIdx.x;
    if (i < n) p[i] = 0.0f;
}

__global__ void sg_scatter_kernel(const int* __restrict__ adj,
                                  const float* __restrict__ val,
                                  const float* __restrict__ support,
                                  float* __restrict__ xg)
{
    long idx = (long)blockIdx.x * blockDim.x + threadIdx.x;
    if (idx >= (long)NNZ_SZ * D_SZ) return;
    int e = (int)(idx >> 7);
    int d = (int)(idx & 127);
    int dst = adj[e];
    int src = adj[NNZ_SZ + e];
    atomicAdd(&xg[(long)dst * D_SZ + d], val[e] * support[(long)src * D_SZ + d]);
}

// hidden = emb[items] + x[items]  (f32 for GRU + bf16 mirror for 3 GEMMs)
__global__ void sg_gather_kernel(const int* __restrict__ items,
                                 const float* __restrict__ emb,
                                 const float* __restrict__ xg,
                                 float* __restrict__ hidden,
                                 unsigned short* __restrict__ hidden_b)
{
    long idx = (long)blockIdx.x * blockDim.x + threadIdx.x;
    if (idx >= (long)NTOK * D_SZ) return;
    long tok = idx >> 7;
    int  d   = (int)(idx & 127);
    int  it  = items[tok];
    float v  = emb[(long)it * D_SZ + d] + xg[(long)it * D_SZ + d];
    hidden[idx]   = v;
    hidden_b[idx] = f32_to_bf16u(v);
}

// inputs[b,l] = [A_in@h_in + b_iah, A_out@h_out + b_oah]  -> bf16 (GEMM-only use)
__global__ __launch_bounds__(128)
void sg_abmm_kernel(const float* __restrict__ Amat,
                    const float* __restrict__ h_in,
                    const float* __restrict__ h_out,
                    const float* __restrict__ b_iah,
                    const float* __restrict__ b_oah,
                    unsigned short* __restrict__ inputs_b)
{
    int bl = blockIdx.x;            // b*L + l
    int b  = bl / L_SZ;
    int d  = threadIdx.x;
    __shared__ float arow[2 * L_SZ];
    if (d < 2 * L_SZ) arow[d] = Amat[(long)bl * (2 * L_SZ) + d];
    __syncthreads();
    float accL = b_iah[d], accR = b_oah[d];
    long hb = (long)b * L_SZ * D_SZ;
    for (int m = 0; m < L_SZ; ++m) {
        accL += arow[m]        * h_in [hb + m * D_SZ + d];
        accR += arow[L_SZ + m] * h_out[hb + m * D_SZ + d];
    }
    inputs_b[(long)bl * 256 + d]       = f32_to_bf16u(accL);
    inputs_b[(long)bl * 256 + 128 + d] = f32_to_bf16u(accR);
}

__global__ void sg_gru_pos_kernel(const float* __restrict__ gi,
                                  const float* __restrict__ gh,
                                  const float* __restrict__ hidden,
                                  float* __restrict__ seq)
{
    long idx = (long)blockIdx.x * blockDim.x + threadIdx.x;
    if (idx >= (long)NTOK * D_SZ) return;
    int  d   = (int)(idx & 127);
    long tok = idx >> 7;
    int  l   = (int)(tok % L_SZ);
    long gb  = tok * 384;
    float r = 1.0f / (1.0f + __expf(-(gi[gb + d]       + gh[gb + d])));
    float z = 1.0f / (1.0f + __expf(-(gi[gb + 128 + d] + gh[gb + 128 + d])));
    float n = tanhf(gi[gb + 256 + d] + r * gh[gb + 256 + d]);
    float hprev = hidden[idx];
    float hnew  = n + z * (hprev - n);
    int i2 = d >> 1;
    float div = __expf((float)(2 * i2) * (-9.210340371976184f / 128.0f));
    float ang = (float)l * div;
    float pe  = (d & 1) ? __cosf(ang) : __sinf(ang);
    seq[idx] = hnew + pe;
}

// LayerNorm: f32 out (residual/readout) + optional bf16 out (GEMM A)
__global__ __launch_bounds__(128)
void sg_layernorm_kernel(const float* __restrict__ x,
                         const float* __restrict__ s,
                         const float* __restrict__ bpar,
                         float* __restrict__ y,
                         unsigned short* __restrict__ yb)
{
    __shared__ float red[128];
    long t = blockIdx.x;
    int  d = threadIdx.x;
    float v = x[t * 128 + d];
    red[d] = v; __syncthreads();
    for (int off = 64; off; off >>= 1) { if (d < off) red[d] += red[d + off]; __syncthreads(); }
    float mean = red[0] * (1.0f / 128.0f);
    __syncthreads();
    float dv = v - mean;
    red[d] = dv * dv; __syncthreads();
    for (int off = 64; off; off >>= 1) { if (d < off) red[d] += red[d + off]; __syncthreads(); }
    float var = red[0] * (1.0f / 128.0f);
    float o = dv * rsqrtf(var + 1e-5f) * s[d] + bpar[d];
    y[t * 128 + d] = o;
    if (yb) yb[t * 128 + d] = f32_to_bf16u(o);
}

// fused small attention; output bf16 (consumed only by out-proj GEMM)
__global__ __launch_bounds__(256)
void sg_attention_kernel(const float* __restrict__ qkv, unsigned short* __restrict__ o)
{
    const int b = blockIdx.x >> 1;
    const int h = blockIdx.x & 1;
    __shared__ float qs[L_SZ][64], ks[L_SZ][64], vs[L_SZ][64];
    __shared__ float att_s[8][52];
    const int tid = threadIdx.x;
    for (int i = tid; i < L_SZ * 64; i += 256) {
        int t = i >> 6, k = i & 63;
        long base = ((long)(b * L_SZ + t)) * 384 + h * 64 + k;
        qs[t][k] = qkv[base];
        ks[t][k] = qkv[base + 128];
        vs[t][k] = qkv[base + 256];
    }
    __syncthreads();
    const int wave = tid >> 5, lane = tid & 31;
    for (int l = wave; l < L_SZ; l += 8) {
        int m0 = lane, m1 = lane + 32;
        float a0 = 0.0f, a1 = 0.0f;
        if (m0 < L_SZ) { float s = 0; for (int k = 0; k < 64; ++k) s += qs[l][k] * ks[m0][k]; a0 = s * 0.125f; }
        if (m1 < L_SZ) { float s = 0; for (int k = 0; k < 64; ++k) s += qs[l][k] * ks[m1][k]; a1 = s * 0.125f; }
        float mx = fmaxf(m0 < L_SZ ? a0 : -1e30f, m1 < L_SZ ? a1 : -1e30f);
        for (int off = 16; off; off >>= 1) mx = fmaxf(mx, __shfl_xor(mx, off, 32));
        float x0 = (m0 < L_SZ) ? __expf(a0 - mx) : 0.0f;
        float x1 = (m1 < L_SZ) ? __expf(a1 - mx) : 0.0f;
        float sm = x0 + x1;
        for (int off = 16; off; off >>= 1) sm += __shfl_xor(sm, off, 32);
        float inv = 1.0f / sm;
        if (m0 < L_SZ) att_s[wave][m0] = x0 * inv;
        if (m1 < L_SZ) att_s[wave][m1] = x1 * inv;
        float o0 = 0.0f, o1 = 0.0f;
        for (int m = 0; m < L_SZ; ++m) {
            float w = att_s[wave][m];
            o0 += w * vs[m][lane];
            o1 += w * vs[m][lane + 32];
        }
        long ob = ((long)(b * L_SZ + l)) * 128 + h * 64;
        o[ob + lane]      = f32_to_bf16u(o0);
        o[ob + lane + 32] = f32_to_bf16u(o1);
    }
}

// attention-pooling readout -> bf16 [a, ht] (consumed only by wt GEMM)
__global__ __launch_bounds__(128)
void sg_readout_kernel(const float* __restrict__ q1,
                       const float* __restrict__ q2,
                       const float* __restrict__ w3,
                       const float* __restrict__ seqn,
                       const float* __restrict__ mask,
                       unsigned short* __restrict__ acat_b)
{
    int b = blockIdx.x, d = threadIdx.x;
    __shared__ float red[128];
    float acc = 0.0f;
    float q1v = q1[(long)b * 128 + d];
    float w3v = w3[d];
    for (int l = 0; l < L_SZ; ++l) {
        long tok = (long)b * L_SZ + l;
        float t = 1.0f / (1.0f + __expf(-(q1v + q2[tok * 128 + d])));
        red[d] = t * w3v;
        __syncthreads();
        for (int off = 64; off; off >>= 1) { if (d < off) red[d] += red[d + off]; __syncthreads(); }
        float alpha = red[0];
        __syncthreads();
        acc += alpha * mask[(long)b * L_SZ + l] * seqn[tok * 128 + d];
    }
    acat_b[(long)b * 256 + d]       = f32_to_bf16u(acc);
    acat_b[(long)b * 256 + 128 + d] = f32_to_bf16u(seqn[((long)b * L_SZ + L_SZ - 1) * 128 + d]);
}

// ---------------------------------------------------------------------------
static void launch_gemm(const unsigned short* Ab, int lda, const unsigned short* Bb,
                        const float* bias, const float* resid,
                        float* C, unsigned short* Cb,
                        int M, int N, int K, int act, hipStream_t s)
{
    dim3 g((N + BN - 1) / BN, (M + BM - 1) / BM);
    sg_gemm_bf16_wmma<<<g, 256, 0, s>>>(Ab, lda, Bb, bias, resid, C, Cb, M, N, K, act);
}

extern "C" void kernel_launch(void* const* d_in, const int* in_sizes, int n_in,
                              void* d_out, int out_size, void* d_ws, size_t ws_size,
                              hipStream_t stream)
{
    const int*   items   = (const int*)  d_in[0];
    const float* Amat    = (const float*)d_in[1];
    const float* mask    = (const float*)d_in[2];
    const int*   adj     = (const int*)  d_in[3];
    const float* adj_val = (const float*)d_in[4];
    const float* emb     = (const float*)d_in[5];
    const float* W0      = (const float*)d_in[6];
    const float* w_ih    = (const float*)d_in[7];
    const float* w_hh    = (const float*)d_in[8];
    const float* b_ih    = (const float*)d_in[9];
    const float* b_hh    = (const float*)d_in[10];
    const float* b_iah   = (const float*)d_in[11];
    const float* b_oah   = (const float*)d_in[12];
    const float* we_in   = (const float*)d_in[13];
    const float* be_in   = (const float*)d_in[14];
    const float* we_out  = (const float*)d_in[15];
    const float* be_out  = (const float*)d_in[16];
    const float* fn_s    = (const float*)d_in[41];
    const float* fn_b    = (const float*)d_in[42];
    const float* w1      = (const float*)d_in[43];
    const float* b1      = (const float*)d_in[44];
    const float* w2      = (const float*)d_in[45];
    const float* b2      = (const float*)d_in[46];
    const float* wt      = (const float*)d_in[48];
    const float* bt      = (const float*)d_in[49];

    float* out = (float*)d_out;

    // --- workspace: f32 activations ---
    float* ws = (float*)d_ws;
    size_t off = 0;
    auto alloc = [&](size_t n) { float* p = ws + off; off += n; return p; };
    float* support = alloc((size_t)NNODE * D_SZ);
    float* xg      = alloc((size_t)NNODE * D_SZ);
    float* hidden  = alloc((size_t)NTOK * D_SZ);
    float* tA      = alloc((size_t)NTOK * 384);   // gi / qkv
    float* tB      = alloc((size_t)NTOK * 384);   // gh
    float* tD      = alloc((size_t)NTOK * D_SZ);  // h_in / xn / seq_n (f32)
    float* tE      = alloc((size_t)NTOK * D_SZ);  // h_out / q2 (f32)
    float* seq     = alloc((size_t)NTOK * D_SZ);
    float* q1buf   = alloc((size_t)B_SZ * D_SZ);

    // --- bf16 region: weight mirrors + bf16 activations ---
    unsigned short* wb = (unsigned short*)(ws + off);
    size_t boff = 0;
    auto balloc = [&](size_t n) { unsigned short* p = wb + boff;
                                  boff += (n + 7) & ~(size_t)7; return p; };
    auto conv = [&](const float* src, unsigned short* dst, long n) {
        int g = (int)((n + 8 * 256 - 1) / (8 * 256));
        sg_f32_to_bf16_kernel<<<g, 256, 0, stream>>>(src, dst, n);
    };
    unsigned short* emb_b   = balloc((size_t)NNODE * D_SZ);
    unsigned short* W0_b    = balloc(D_SZ * D_SZ);
    unsigned short* wein_b  = balloc(D_SZ * D_SZ);
    unsigned short* weout_b = balloc(D_SZ * D_SZ);
    unsigned short* wih_b   = balloc(3 * D_SZ * 2 * D_SZ);
    unsigned short* whh_b   = balloc(3 * D_SZ * D_SZ);
    unsigned short* lyr_b[2][4];
    for (int li = 0; li < 2; ++li) {
        lyr_b[li][0] = balloc(3 * D_SZ * D_SZ);   // in_w
        lyr_b[li][1] = balloc(D_SZ * D_SZ);       // out_w
        lyr_b[li][2] = balloc(FF_SZ * D_SZ);      // l1_w
        lyr_b[li][3] = balloc(D_SZ * FF_SZ);      // l2_w
    }
    unsigned short* w1_b = balloc(D_SZ * D_SZ);
    unsigned short* w2_b = balloc(D_SZ * D_SZ);
    unsigned short* wt_b = balloc(D_SZ * 2 * D_SZ);
    // bf16 activations
    unsigned short* hidden_b = balloc((size_t)NTOK * D_SZ);
    unsigned short* tC_b     = balloc((size_t)NTOK * 256);   // inputs / ff1
    unsigned short* tD_b     = balloc((size_t)NTOK * D_SZ);  // xn / seq_n
    unsigned short* tE_b     = balloc((size_t)NTOK * D_SZ);  // attn out
    unsigned short* acat_b   = balloc((size_t)B_SZ * 256);
    unsigned short* at2_b    = balloc((size_t)B_SZ * D_SZ);
    (void)ws_size; (void)n_in; (void)in_sizes; (void)out_size;

    // 0) weight bf16 mirrors
    conv(emb,    emb_b,   (long)NNODE * D_SZ);
    conv(W0,     W0_b,    D_SZ * D_SZ);
    conv(we_in,  wein_b,  D_SZ * D_SZ);
    conv(we_out, weout_b, D_SZ * D_SZ);
    conv(w_ih,   wih_b,   3 * D_SZ * 2 * D_SZ);
    conv(w_hh,   whh_b,   3 * D_SZ * D_SZ);
    for (int li = 0; li < 2; ++li) {
        int base = 17 + li * 12;
        conv((const float*)d_in[base + 0], lyr_b[li][0], 3 * D_SZ * D_SZ);
        conv((const float*)d_in[base + 2], lyr_b[li][1], D_SZ * D_SZ);
        conv((const float*)d_in[base + 4], lyr_b[li][2], FF_SZ * D_SZ);
        conv((const float*)d_in[base + 6], lyr_b[li][3], D_SZ * FF_SZ);
    }
    conv(w1, w1_b, D_SZ * D_SZ);
    conv(w2, w2_b, D_SZ * D_SZ);
    conv(wt, wt_b, D_SZ * 2 * D_SZ);

    // 1) GCN: support = emb @ W0^T   (A = bf16 emb mirror)
    launch_gemm(emb_b, D_SZ, W0_b, nullptr, nullptr, support, nullptr,
                NNODE, D_SZ, D_SZ, 0, stream);

    // 2) segment-sum scatter
    long nz = (long)NNODE * D_SZ;
    sg_zero_kernel<<<(int)((nz + 255) / 256), 256, 0, stream>>>(xg, nz);
    long sc = (long)NNZ_SZ * D_SZ;
    sg_scatter_kernel<<<(int)((sc + 255) / 256), 256, 0, stream>>>(adj, adj_val, support, xg);

    // 3) hidden = emb[items] + x[items]
    long nt = (long)NTOK * D_SZ;
    sg_gather_kernel<<<(int)((nt + 255) / 256), 256, 0, stream>>>(items, emb, xg,
                                                                  hidden, hidden_b);

    // 4) GNN cell (STEP=1)
    launch_gemm(hidden_b, D_SZ, wein_b,  be_in,  nullptr, tD, nullptr, NTOK, D_SZ, D_SZ, 0, stream);
    launch_gemm(hidden_b, D_SZ, weout_b, be_out, nullptr, tE, nullptr, NTOK, D_SZ, D_SZ, 0, stream);
    sg_abmm_kernel<<<NTOK, 128, 0, stream>>>(Amat, tD, tE, b_iah, b_oah, tC_b);
    launch_gemm(tC_b, 256, wih_b, b_ih, nullptr, tA, nullptr, NTOK, 3 * D_SZ, 2 * D_SZ, 0, stream);
    launch_gemm(hidden_b, D_SZ, whh_b, b_hh, nullptr, tB, nullptr, NTOK, 3 * D_SZ, D_SZ, 0, stream);
    sg_gru_pos_kernel<<<(int)((nt + 255) / 256), 256, 0, stream>>>(tA, tB, hidden, seq);

    // 5) transformer encoder layers
    for (int li = 0; li < 2; ++li) {
        int base = 17 + li * 12;
        const float* in_b_  = (const float*)d_in[base + 1];
        const float* out_b_ = (const float*)d_in[base + 3];
        const float* l1_b_  = (const float*)d_in[base + 5];
        const float* l2_b_  = (const float*)d_in[base + 7];
        const float* n1_s   = (const float*)d_in[base + 8];
        const float* n1_b   = (const float*)d_in[base + 9];
        const float* n2_s   = (const float*)d_in[base + 10];
        const float* n2_b   = (const float*)d_in[base + 11];

        sg_layernorm_kernel<<<NTOK, 128, 0, stream>>>(seq, n1_s, n1_b, tD, tD_b);
        launch_gemm(tD_b, D_SZ, lyr_b[li][0], in_b_, nullptr, tA, nullptr,
                    NTOK, 3 * D_SZ, D_SZ, 0, stream);
        sg_attention_kernel<<<B_SZ * 2, 256, 0, stream>>>(tA, tE_b);
        launch_gemm(tE_b, D_SZ, lyr_b[li][1], out_b_, seq, seq, nullptr,
                    NTOK, D_SZ, D_SZ, 0, stream);
        sg_layernorm_kernel<<<NTOK, 128, 0, stream>>>(seq, n2_s, n2_b, tD, tD_b);
        launch_gemm(tD_b, D_SZ, lyr_b[li][2], l1_b_, nullptr, nullptr, tC_b,
                    NTOK, FF_SZ, D_SZ, 1, stream);                 // relu -> bf16
        launch_gemm(tC_b, FF_SZ, lyr_b[li][3], l2_b_, seq, seq, nullptr,
                    NTOK, D_SZ, FF_SZ, 0, stream);
    }

    // 6) final layernorm, attention-pooling readout
    sg_layernorm_kernel<<<NTOK, 128, 0, stream>>>(seq, fn_s, fn_b, tD, tD_b);
    launch_gemm(tD_b + (L_SZ - 1) * D_SZ, L_SZ * D_SZ, w1_b, b1, nullptr, q1buf, nullptr,
                B_SZ, D_SZ, D_SZ, 0, stream);                      // ht @ w1^T (strided A)
    launch_gemm(tD_b, D_SZ, w2_b, b2, nullptr, tE, nullptr, NTOK, D_SZ, D_SZ, 0, stream);
    sg_readout_kernel<<<B_SZ, 128, 0, stream>>>(q1buf, tE, (const float*)d_in[47],
                                                tD, mask, acat_b);

    // 7) a = [a, ht] @ wt^T + bt  -> bf16 only (scoring A)
    launch_gemm(acat_b, 256, wt_b, bt, nullptr, nullptr, at2_b,
                B_SZ, D_SZ, 2 * D_SZ, 0, stream);

    // 8) scores = a @ emb[1:]^T -> (1024, 49999)
    launch_gemm(at2_b, D_SZ, emb_b + D_SZ, nullptr, nullptr, out, nullptr,
                B_SZ, NNODE - 1, D_SZ, 0, stream);
}
